// MultiHeadAttention_22960895164825
// MI455X (gfx1250) — matrix operand order repaired and checked by reading, not compiled, not run on hardware
//
#include <hip/hip_runtime.h>
#include <hip/hip_bf16.h>
#include <stdint.h>

// ---------------------------------------------------------------------------
// Linear attention (phi = silu+1), B=8, C=1024, H=16, DH=64, Tq=Tk=1500.
// Heavy work = 4 GEMMs (1024x1024)x(1024x1500) per batch, ~101 GFLOP.
// bf16 A/B + f32 accumulate via v_wmma_f32_16x16x32_bf16 (wave32).
// Each wave register-blocks a 64x32 output tile: 8 WMMAs per 12 b128 loads.
// ---------------------------------------------------------------------------

typedef __attribute__((ext_vector_type(16))) __bf16 v16bf;
typedef __attribute__((ext_vector_type(8)))  float  v8f;
typedef __attribute__((ext_vector_type(4)))  unsigned int v4u;

typedef unsigned short bfbits;   // raw bf16 bit storage

union FragCvt {
    struct { v4u lo; v4u hi; } u;
    v16bf b;
};

__device__ __forceinline__ bfbits f2bf(float f) {
    unsigned int u = __float_as_uint(f);
    unsigned int r = (u + 0x7FFFu + ((u >> 16) & 1u)) >> 16;   // RNE
    return (bfbits)r;
}

// --------------------------- f32 -> bf16 copy ------------------------------
__global__ void __launch_bounds__(256)
k_cvt_bf16(const float* __restrict__ src, bfbits* __restrict__ dst, int n) {
    int i = blockIdx.x * 256 + threadIdx.x;
    if (i < n) dst[i] = f2bf(src[i]);
}

// ------------- x (B,C,T) f32  ->  xT (B,Tpad,C) bf16 (zero-padded) ---------
__global__ void __launch_bounds__(256)
k_transpose_cvt(const float* __restrict__ x, bfbits* __restrict__ xT,
                int Bn, int C, int T, int Tpad) {
    size_t idx = (size_t)blockIdx.x * 256 + threadIdx.x;
    size_t total = (size_t)Bn * C * Tpad;
    if (idx >= total) return;
    int t = (int)(idx % Tpad);
    size_t bc = idx / Tpad;
    int c = (int)(bc % C);
    int b = (int)(bc / C);
    float v = (t < T) ? x[((size_t)b * C + c) * T + t] : 0.0f;
    xT[((size_t)b * Tpad + t) * C + c] = f2bf(v);
}

// --------------------------- WMMA bf16 GEMM --------------------------------
// Y[b][m][t] = sum_k W[m][k] * XT[b][t][k]   (+bias, optional phi)
// Wave computes a 64x32 block = 4x2 tiles of 16x16; K-loop steps by 32.
#define WM 4
#define WN 2
__global__ void __launch_bounds__(256)
k_gemm_bf16(const bfbits* __restrict__ W,    // M x K row-major
            const bfbits* __restrict__ XT,   // per batch: Tpad x K row-major
            float* __restrict__ Y,           // per batch: M x ldY
            const float* __restrict__ bias,  // M floats or nullptr
            int M, int K, int Tpad, int ldY, int Nvalid, int nBatch,
            int applyPhi) {
    const int wave = blockIdx.x * (blockDim.x >> 5) + (threadIdx.x >> 5);
    const int lane = threadIdx.x & 31;
    const int Mb = M / (16 * WM);            // wave-blocks along M
    const int Nb = Tpad / (16 * WN);         // wave-blocks along N
    const int perBatch = Mb * Nb;
    if (wave >= perBatch * nBatch) return;

    const int b = wave / perBatch;
    const int r = wave - b * perBatch;
    const int mBlk = r / Nb;
    const int nBlk = r - mBlk * Nb;
    const int mBase = mBlk * (16 * WM);
    const int nBase = nBlk * (16 * WN);

    const int laneLo = lane & 15;
    const int half   = lane >> 4;            // 0 or 1

    // A: row (mBase + 16*i + laneLo) of W; K-halves per ISA 16-bit A layout.
    const bfbits* Wrow = W + (size_t)(mBase + laneLo) * K + half * 8;
    // B: column t = (nBase + 16*j + laneLo) of X == row of XT.
    const bfbits* Xrow = XT + ((size_t)b * Tpad + nBase + laneLo) * K + half * 8;
    const size_t aStride = (size_t)16 * K;   // next M-tile
    const size_t bStride = (size_t)16 * K;   // next N-tile

    v8f acc[WM * WN];
#pragma unroll
    for (int i = 0; i < WM * WN; ++i) acc[i] = (v8f){};

    for (int k = 0; k < K; k += 32) {
        FragCvt a[WM], bb[WN];
#pragma unroll
        for (int i = 0; i < WM; ++i) {
            a[i].u.lo = *(const v4u*)(Wrow + i * aStride + k);        // K=half*8+0..7
            a[i].u.hi = *(const v4u*)(Wrow + i * aStride + k + 16);   // K=16+half*8+0..7
        }
#pragma unroll
        for (int j = 0; j < WN; ++j) {
            bb[j].u.lo = *(const v4u*)(Xrow + j * bStride + k);       // K=half*8+0..7
            bb[j].u.hi = *(const v4u*)(Xrow + j * bStride + k + 16);  // K=16+half*8+0..7
        }
        __builtin_prefetch((const void*)(Xrow + k + 128), 0, 1);
#pragma unroll
        for (int i = 0; i < WM; ++i)
#pragma unroll
            for (int j = 0; j < WN; ++j)
                acc[i * WN + j] = __builtin_amdgcn_wmma_f32_16x16x32_bf16(
                    false, a[i].b, false, bb[j].b, (short)0, acc[i * WN + j],
                    false, false);
    }

    float* Yb = Y + (size_t)b * M * ldY;
#pragma unroll
    for (int i = 0; i < WM; ++i) {
#pragma unroll
        for (int j = 0; j < WN; ++j) {
            const int col = nBase + j * 16 + laneLo;     // N = lane&15
            const bool ok = (col < Nvalid);
            v8f v = acc[i * WN + j];
#pragma unroll
            for (int e = 0; e < 8; ++e) {
                const int m = mBase + i * 16 + e + half * 8;  // M = e + 8*(lane>=16)
                float val = v[e];
                if (bias) val += bias[m];
                if (applyPhi) val = val / (1.0f + __expf(-val)) + 1.0f; // silu+1
                if (ok) Yb[(size_t)m * ldY + col] = val;
            }
        }
    }
}

// ------------- per-channel reductions: k_sum, kv_sum over t ----------------
__global__ void __launch_bounds__(256)
k_reduce(const float* __restrict__ pk, const float* __restrict__ vv,
         float* __restrict__ ksum, float* __restrict__ kvsum,
         int Tpad, int Tk, int totalRows) {
    int row = blockIdx.x * (blockDim.x >> 5) + (threadIdx.x >> 5);
    int lane = threadIdx.x & 31;
    if (row >= totalRows) return;
    const float* pr = pk + (size_t)row * Tpad;
    const float* vr = vv + (size_t)row * Tpad;
    float s = 0.0f, sv = 0.0f;
    for (int t = lane; t < Tk; t += 32) {
        float p = pr[t];
        s  += p;
        sv += p * vr[t];
    }
    for (int off = 16; off > 0; off >>= 1) {
        s  += __shfl_down(s,  off, 32);
        sv += __shfl_down(sv, off, 32);
    }
    if (lane == 0) { ksum[row] = s; kvsum[row] = sv; }
}

// ---- out = pq * kv_sum / (pq . k_sum + eps) / sqrt(DH), store bf16 T-major -
__global__ void __launch_bounds__(256)
k_normalize(const float* __restrict__ pq, const float* __restrict__ ksum,
            const float* __restrict__ kvsum, bfbits* __restrict__ outT,
            int C, int Tpad, int Tq) {
    __shared__ float ks[64];
    __shared__ float kv[64];
    const int bh = blockIdx.y;           // b*16 + h
    const int b = bh >> 4, h = bh & 15;
    const int rowBase = b * C + h * 64;
    if (threadIdx.x < 64) {
        ks[threadIdx.x] = ksum[rowBase + threadIdx.x];
        kv[threadIdx.x] = kvsum[rowBase + threadIdx.x];
    }
    __syncthreads();
    const int t = blockIdx.x * 256 + threadIdx.x;
    if (t >= Tq) return;
    const float* p = pq + (size_t)rowBase * Tpad + t;
    float den = 1e-6f;
    for (int d = 0; d < 64; ++d) den += p[(size_t)d * Tpad] * ks[d];
    const float inv = 1.0f / (den * 8.0f);          // 8 = sqrt(DH)
    bfbits* o = outT + ((size_t)b * Tpad + t) * C + h * 64;
    for (int d = 0; d < 64; ++d) o[d] = f2bf(p[(size_t)d * Tpad] * kv[d] * inv);
}

// ---------------------------------------------------------------------------
static inline size_t alignUp(size_t x) { return (x + 255) & ~(size_t)255; }

extern "C" void kernel_launch(void* const* d_in, const int* in_sizes, int n_in,
                              void* d_out, int out_size, void* d_ws, size_t ws_size,
                              hipStream_t stream) {
    (void)n_in; (void)out_size; (void)ws_size;
    const float* x  = (const float*)d_in[0];
    const float* xa = (const float*)d_in[1];
    const float* Wq = (const float*)d_in[2];
    const float* bq = (const float*)d_in[3];
    const float* Wk = (const float*)d_in[4];
    const float* Wv = (const float*)d_in[5];
    const float* bv = (const float*)d_in[6];
    const float* Wo = (const float*)d_in[7];
    const float* bo = (const float*)d_in[8];

    const int B = 8, C = 1024, H = 16;
    const int Tq = in_sizes[0] / (B * C);
    const int Tk = in_sizes[1] / (B * C);
    const int Tqp = (Tq + 31) & ~31;     // 1504 (multiple of 32 for WN=2)
    const int Tkp = (Tk + 31) & ~31;

    // ---- workspace layout ----
    char* p = (char*)d_ws;
    const size_t wBytes  = (size_t)C * C * sizeof(bfbits);
    const size_t xTqB    = (size_t)B * Tqp * C * sizeof(bfbits);
    const size_t xTkB    = (size_t)B * Tkp * C * sizeof(bfbits);
    const size_t actQB   = (size_t)B * C * Tqp * sizeof(float);
    const size_t actKB   = (size_t)B * C * Tkp * sizeof(float);

    bfbits* Wq_b = (bfbits*)p; p += alignUp(wBytes);
    bfbits* Wk_b = (bfbits*)p; p += alignUp(wBytes);
    bfbits* Wv_b = (bfbits*)p; p += alignUp(wBytes);
    bfbits* Wo_b = (bfbits*)p; p += alignUp(wBytes);
    bfbits* xT   = (bfbits*)p; p += alignUp(xTqB);   // reused later as outT
    bfbits* xaT  = (bfbits*)p; p += alignUp(xTkB);
    float*  pq   = (float*)p;  p += alignUp(actQB);
    float*  pk   = (float*)p;  p += alignUp(actKB);
    float*  vv   = (float*)p;  p += alignUp(actKB);
    float*  ksum = (float*)p;  p += alignUp((size_t)B * C * sizeof(float));
    float*  kvsum= (float*)p;  p += alignUp((size_t)B * C * sizeof(float));
    bfbits* outT = xT;                                // xT dead after q-GEMM

    // ---- 1) weights -> bf16 ----
    const int nW = C * C;
    k_cvt_bf16<<<(nW + 255) / 256, 256, 0, stream>>>(Wq, Wq_b, nW);
    k_cvt_bf16<<<(nW + 255) / 256, 256, 0, stream>>>(Wk, Wk_b, nW);
    k_cvt_bf16<<<(nW + 255) / 256, 256, 0, stream>>>(Wv, Wv_b, nW);
    k_cvt_bf16<<<(nW + 255) / 256, 256, 0, stream>>>(Wo, Wo_b, nW);

    // ---- 2) activations -> transposed bf16 ----
    {
        size_t tot = (size_t)B * C * Tqp;
        k_transpose_cvt<<<(unsigned)((tot + 255) / 256), 256, 0, stream>>>(x, xT, B, C, Tq, Tqp);
        tot = (size_t)B * C * Tkp;
        k_transpose_cvt<<<(unsigned)((tot + 255) / 256), 256, 0, stream>>>(xa, xaT, B, C, Tk, Tkp);
    }

    // ---- 3) projections (WMMA) ----
    const int wavesQ = (C / (16 * WM)) * (Tqp / (16 * WN)) * B;
    const int wavesK = (C / (16 * WM)) * (Tkp / (16 * WN)) * B;
    const int blkQ = (wavesQ + 7) / 8, blkK = (wavesK + 7) / 8;
    // q = phi(Wq x + bq)
    k_gemm_bf16<<<blkQ, 256, 0, stream>>>(Wq_b, xT,  pq, bq, C, C, Tqp, Tqp, Tqp, B, 1);
    // k = phi(Wk xa)      (no bias)
    k_gemm_bf16<<<blkK, 256, 0, stream>>>(Wk_b, xaT, pk, nullptr, C, C, Tkp, Tkp, Tkp, B, 1);
    // v = Wv xa + bv
    k_gemm_bf16<<<blkK, 256, 0, stream>>>(Wv_b, xaT, vv, bv, C, C, Tkp, Tkp, Tkp, B, 0);

    // ---- 4) k_sum / kv_sum reductions ----
    const int rows = B * C;
    k_reduce<<<(rows + 7) / 8, 256, 0, stream>>>(pk, vv, ksum, kvsum, Tkp, Tk, rows);

    // ---- 5) normalize + write bf16 transposed ----
    dim3 gN((Tq + 255) / 256, B * H);
    k_normalize<<<gN, 256, 0, stream>>>(pq, ksum, kvsum, outT, C, Tqp, Tq);

    // ---- 6) y = Wo out + bo  -> d_out (B,C,Tq,1), guarded stores t<Tq ----
    k_gemm_bf16<<<blkQ, 256, 0, stream>>>(Wo_b, outT, (float*)d_out, bo,
                                          C, C, Tqp, Tq, Tq, B, 0);
}
